// TransformerBlock_35012573397635
// MI455X (gfx1250) — matrix operand order repaired and checked by reading, not compiled, not run on hardware
//
#include <hip/hip_runtime.h>
#include <hip/hip_bf16.h>
#include <stdint.h>

// Problem constants (fixed by the reference)
#define BB 2
#define NN 1024
#define MM 1024
#define HH 16
#define DD 64
#define DIMN 1024
#define JJ (MM + NN)
#define WIN 256

typedef __attribute__((ext_vector_type(16))) __bf16 bf16x16;
typedef __attribute__((ext_vector_type(8)))  float  f32x8;
typedef __attribute__((ext_vector_type(4)))  unsigned int u32x4;
typedef __attribute__((ext_vector_type(8)))  int i32x8;
typedef __attribute__((ext_vector_type(4)))  int i32x4;

union ABfrag { bf16x16 v; unsigned int u[8]; uint4 q[2]; };

__device__ __forceinline__ __bf16 tobf(float f) { return (__bf16)f; }

// ---------------------------------------------------------------------------
// Tensor Data Mover availability (device pass only; host pass parses fallback)
// ---------------------------------------------------------------------------
#ifndef __has_builtin
#define __has_builtin(x) 0
#endif
#if __has_builtin(__builtin_amdgcn_tensor_load_to_lds) && \
    __has_builtin(__builtin_amdgcn_s_wait_tensorcnt)
#define USE_TDM 1
#else
#define USE_TDM 0
#endif

#if USE_TDM
// 2D tile load global->LDS via TDM, bf16 elements (data_size=1 -> 2B).
// LDS padding: 4 DWORDs after every 16 DWORDs => row stride 40 bf16 (80 B),
// matching the manual LDS layout used by the fragment gathers.
__device__ __forceinline__ void tdm_load_tile(
    unsigned int lds_off, const __bf16* gaddr, int K_elems, int rows, int tile_rows) {
  uint64_t ga = (uint64_t)(uintptr_t)gaddr;
  u32x4 g0 = { 1u,                                   // count=1 (valid D#)
               lds_off,                               // LDS byte address
               (unsigned int)ga,
               (unsigned int)((ga >> 32) & 0x1FFFFFFu) | (2u << 30) }; // type=2
  unsigned int flags = (1u << 16)   // data_size = 2 bytes
                     | (1u << 20)   // pad_enable
                     | (3u << 22)   // pad_interval: 16 DWORDs
                     | (3u << 25);  // pad_amount:   4 DWORDs
  i32x8 g1 = { (int)flags,
               (int)(((unsigned)K_elems & 0xFFFFu) << 16),          // tensor_dim0 lo16
               (int)((((unsigned)K_elems) >> 16) |
                     (((unsigned)rows & 0xFFFFu) << 16)),           // dim0 hi / dim1 lo
               (int)((((unsigned)rows) >> 16) | (32u << 16)),       // dim1 hi / tile_dim0=32
               (int)(unsigned)tile_rows,                            // tile_dim1 (tile_dim2=0)
               (int)(unsigned)K_elems,                              // tensor_dim0_stride lo32
               0, 0 };
  i32x4 gz = { 0, 0, 0, 0 };
#if __clang_major__ >= 23
  i32x8 gz8 = { 0, 0, 0, 0, 0, 0, 0, 0 };
  __builtin_amdgcn_tensor_load_to_lds(g0, g1, gz, gz, gz8, 0);
#else
  __builtin_amdgcn_tensor_load_to_lds(g0, g1, gz, gz, 0);
#endif
}
#endif

// ---------------------------------------------------------------------------
// fp32 [K,N] -> bf16 transposed [N,K] (weights), 32x32 LDS tiles
// ---------------------------------------------------------------------------
__global__ __launch_bounds__(256) void cvt_transpose_kernel(
    const float* __restrict__ in, __bf16* __restrict__ out, int Kd, int Nd) {
  __shared__ float tile[32][33];
  const int kb = blockIdx.x * 32, nb = blockIdx.y * 32;
  const int tx = threadIdx.x & 31, ty = threadIdx.x >> 5;
#pragma unroll
  for (int j = 0; j < 4; j++)
    tile[ty + j * 8][tx] = in[(size_t)(kb + ty + j * 8) * Nd + nb + tx];
  __syncthreads();
#pragma unroll
  for (int j = 0; j < 4; j++)
    out[(size_t)(nb + ty + j * 8) * Kd + kb + tx] = tobf(tile[tx][ty + j * 8]);
}

// ---------------------------------------------------------------------------
// LayerNorm over DIMN=1024, one block (256 thr) per row.
// ---------------------------------------------------------------------------
__global__ __launch_bounds__(256) void ln_kernel(
    const float* __restrict__ A, const float* __restrict__ R,
    const float* __restrict__ g, const float* __restrict__ bta,
    float* __restrict__ hOut, __bf16* __restrict__ outb) {
  const int row = blockIdx.x;
  const int t = threadIdx.x;
  __shared__ float red[256];
  float v[4];
  const size_t base = (size_t)row * DIMN + t * 4;
#pragma unroll
  for (int i = 0; i < 4; i++) {
    v[i] = A[base + i];
    if (R) v[i] += R[base + i];
  }
  if (hOut) {
#pragma unroll
    for (int i = 0; i < 4; i++) hOut[base + i] = v[i];
  }
  float s = v[0] + v[1] + v[2] + v[3];
  red[t] = s; __syncthreads();
  for (int off = 128; off > 0; off >>= 1) { if (t < off) red[t] += red[t + off]; __syncthreads(); }
  const float mu = red[0] * (1.0f / DIMN);
  __syncthreads();
  float sq = 0.f;
#pragma unroll
  for (int i = 0; i < 4; i++) { float d = v[i] - mu; sq += d * d; }
  red[t] = sq; __syncthreads();
  for (int off = 128; off > 0; off >>= 1) { if (t < off) red[t] += red[t + off]; __syncthreads(); }
  const float rs = rsqrtf(red[0] * (1.0f / DIMN) + 1e-5f);
#pragma unroll
  for (int i = 0; i < 4; i++)
    outb[base + i] = tobf((v[i] - mu) * rs * g[t * 4 + i] + bta[t * 4 + i]);
}

// ---------------------------------------------------------------------------
// bf16 WMMA GEMM: C[M,N] = A[M,K] @ Bt[N,K]^T, fp32 accumulate.
// Block tile 128x64, 256 threads (8 waves), each wave owns 32x32 (4 WMMAs/step).
// Double-buffered LDS; tiles staged via TDM (tensor_load_to_lds) when available.
// MODE 0: fp32 out. 1: fp32*scale. 2: bf16 relu(acc+bias). 3: fp32 acc+bias+Res.
// ---------------------------------------------------------------------------
template <int MODE>
__global__ __launch_bounds__(256) void gemm_kernel(
    const __bf16* __restrict__ A, const __bf16* __restrict__ Bt,
    int K, int Nld,
    float* __restrict__ Cf, __bf16* __restrict__ Cb,
    const float* __restrict__ bias, const float* __restrict__ Res, float scale) {
  __shared__ __align__(16) __bf16 sA[2][128 * 40];
  __shared__ __align__(16) __bf16 sB[2][64 * 40];

  const int t = threadIdx.x;
  const int lane = t & 31, w = t >> 5;
  const int m0 = blockIdx.y * 128, n0 = blockIdx.x * 64;
  const int mw = w >> 1;   // 4 M-groups of 32 rows
  const int nw = w & 1;    // 2 N-groups of 32 cols

  f32x8 acc00 = {}, acc01 = {}, acc10 = {}, acc11 = {};

  const int lrow = lane & 15;
  const int kb  = (lane >> 4) * 8;
  const int kb2 = (lane >> 4) * 16;
  const int ar0 = (mw * 32 + lrow) * 40;
  const int ar1 = (mw * 32 + 16 + lrow) * 40;
  const int br0 = (nw * 32 + lrow) * 40;
  const int br1 = (nw * 32 + 16 + lrow) * 40;

  const int KT = K >> 5;

#if !USE_TDM
  // manual staging indices
  const int sar = t >> 1, sac = (t & 1) * 16;  // A: 128 rows x 32, 16 bf16/thread
  const int sbr = t >> 2, sbc = (t & 3) * 8;   // Bt: 64 rows x 32, 8 bf16/thread
  auto stage = [&](int buf, int k0) {
    const __bf16* ag = A + (size_t)(m0 + sar) * K + k0 + sac;
    *(uint4*)&sA[buf][sar * 40 + sac]     = *(const uint4*)ag;
    *(uint4*)&sA[buf][sar * 40 + sac + 8] = *(const uint4*)(ag + 8);
    const __bf16* bg = Bt + (size_t)(n0 + sbr) * K + k0 + sbc;
    *(uint4*)&sB[buf][sbr * 40 + sbc] = *(const uint4*)bg;
    if (k0 + 64 < K) {  // prefetch next-next tile
      __builtin_prefetch(ag + 64, 0, 1);
      __builtin_prefetch(bg + 64, 0, 1);
    }
  };
  stage(0, 0);
  __syncthreads();
#else
  if (w == 0) {
    tdm_load_tile((unsigned int)(uintptr_t)&sA[0][0], A + (size_t)m0 * K, K, 128, 128);
    tdm_load_tile((unsigned int)(uintptr_t)&sB[0][0], Bt + (size_t)n0 * K, K, 64, 64);
  }
#endif

  for (int kt = 0; kt < KT; kt++) {
    const int buf = kt & 1;
#if USE_TDM
    if (w == 0) {
      if (kt + 1 < KT) {
        const int k1 = (kt + 1) << 5;
        tdm_load_tile((unsigned int)(uintptr_t)&sA[buf ^ 1][0],
                      A + (size_t)m0 * K + k1, K, 128, 128);
        tdm_load_tile((unsigned int)(uintptr_t)&sB[buf ^ 1][0],
                      Bt + (size_t)n0 * K + k1, K, 64, 64);
        __builtin_amdgcn_s_wait_tensorcnt(2);  // current buffer's 2 loads done
      } else {
        __builtin_amdgcn_s_wait_tensorcnt(0);
      }
    }
    __syncthreads();
#else
    if (kt + 1 < KT) stage(buf ^ 1, (kt + 1) << 5);
#endif

    const __bf16* As = sA[buf];
    const __bf16* Bs = sB[buf];
    ABfrag a0, a1, b0, b1;
#pragma unroll
    for (int v2 = 0; v2 < 4; v2++) {
      a0.u[v2]     = *(const unsigned*)&As[ar0 + kb + 2 * v2];
      a0.u[4 + v2] = *(const unsigned*)&As[ar0 + kb + 16 + 2 * v2];
      a1.u[v2]     = *(const unsigned*)&As[ar1 + kb + 2 * v2];
      a1.u[4 + v2] = *(const unsigned*)&As[ar1 + kb + 16 + 2 * v2];
    }
    { const __bf16* p = &Bs[br0 + kb]; b0.q[0] = *(const uint4*)p; b0.q[1] = *(const uint4*)(p + 16); }
    { const __bf16* p = &Bs[br1 + kb]; b1.q[0] = *(const uint4*)p; b1.q[1] = *(const uint4*)(p + 16); }

    acc00 = __builtin_amdgcn_wmma_f32_16x16x32_bf16(false, a0.v, false, b0.v, (short)0, acc00, false, false);
    acc01 = __builtin_amdgcn_wmma_f32_16x16x32_bf16(false, a0.v, false, b1.v, (short)0, acc01, false, false);
    acc10 = __builtin_amdgcn_wmma_f32_16x16x32_bf16(false, a1.v, false, b0.v, (short)0, acc10, false, false);
    acc11 = __builtin_amdgcn_wmma_f32_16x16x32_bf16(false, a1.v, false, b1.v, (short)0, acc11, false, false);
    __syncthreads();
  }

  // Epilogue: per 16x16 tile, lane col = lane&15, rows (lane>>4)*8 + r
  const int rb0 = m0 + mw * 32 + (lane >> 4) * 8;
  const int c0 = n0 + nw * 32 + (lane & 15);
  f32x8 accs[4] = { acc00, acc01, acc10, acc11 };
#pragma unroll
  for (int mi = 0; mi < 2; mi++) {
#pragma unroll
    for (int ni = 0; ni < 2; ni++) {
      f32x8 acc = accs[mi * 2 + ni];
      const int col = c0 + ni * 16;
#pragma unroll
      for (int r = 0; r < 8; r++) {
        const size_t ro = (size_t)(rb0 + mi * 16 + r) * Nld;
        float va = acc[r];
        if constexpr (MODE == 0) {
          Cf[ro + col] = va;
        } else if constexpr (MODE == 1) {
          Cf[ro + col] = va * scale;
        } else if constexpr (MODE == 2) {
          Cb[ro + col] = tobf(fmaxf(va + bias[col], 0.f));
        } else {
          Cf[ro + col] = va + bias[col] + Res[ro + col];
        }
      }
    }
  }
}

// ---------------------------------------------------------------------------
// mem_kv_new assembly
// ---------------------------------------------------------------------------
__global__ void copy_memkv_kernel(const float* __restrict__ memkv, float* __restrict__ out) {
  const size_t n = (size_t)BB * HH * MM * 2 * DD;
  for (size_t i = blockIdx.x * (size_t)blockDim.x + threadIdx.x; i < n;
       i += (size_t)gridDim.x * blockDim.x) {
    size_t c = i % (2 * DD);
    size_t m = (i / (2 * DD)) % MM;
    size_t bh = i / ((size_t)2 * DD * MM);
    out[(bh * JJ + m) * (2 * DD) + c] = memkv[i];
  }
}

__global__ void scatter_kv_kernel(const float* __restrict__ kv, float* __restrict__ out) {
  const size_t n = (size_t)BB * NN * HH * 2 * DD;
  for (size_t i = blockIdx.x * (size_t)blockDim.x + threadIdx.x; i < n;
       i += (size_t)gridDim.x * blockDim.x) {
    size_t c = i % (2 * DD);
    size_t h = (i / (2 * DD)) % HH;
    size_t row = i / ((size_t)2 * DD * HH);
    size_t b = row / NN, nq = row % NN;
    out[(((b * HH) + h) * JJ + MM + nq) * (2 * DD) + c] = kv[i];
  }
}

// ---------------------------------------------------------------------------
// Windowed attention, one wave per (b, h, 16-query-row tile).
// ---------------------------------------------------------------------------
__global__ __launch_bounds__(32) void attn_kernel(
    const float* __restrict__ q,       // [B*N, DIM], already scaled by D^-0.5
    const float* __restrict__ memkv,   // [B,H,J,2D] (mem_kv_new in d_out)
    float* __restrict__ attnw,         // [B,N,J] (zeroed)
    float* __restrict__ hattn) {       // [B*N, DIM]
  const int lane = threadIdx.x;
  const int n0 = blockIdx.x * 16;
  const int h = blockIdx.y;
  const int b = blockIdx.z;
  const int jlo = MM + n0 - WIN;  // >= 768, multiple of 16

  __shared__ __align__(16) __bf16 sQ[16 * 72];
  __shared__ __align__(16) __bf16 sK[16 * 72];
  __shared__ __align__(16) __bf16 sV[64 * 40];   // transposed: [d][key32]
  __shared__ float sS[16 * 292];                  // 288 cols used

  for (int i = lane; i < 16 * 64; i += 32) {
    int r = i >> 6, d = i & 63;
    sQ[r * 72 + d] = tobf(q[(size_t)(b * NN + n0 + r) * DIMN + h * DD + d]);
  }
  __syncthreads();

  const int kb = (lane >> 4) * 8;
  const int kb2 = (lane >> 4) * 16;
  const int qr = lane & 15;

  ABfrag aq0, aq1;  // d 0..31 and 32..63
#pragma unroll
  for (int v2 = 0; v2 < 4; v2++) {
    aq0.u[v2]     = *(const unsigned*)&sQ[qr * 72 + kb + 2 * v2];
    aq0.u[4 + v2] = *(const unsigned*)&sQ[qr * 72 + kb + 16 + 2 * v2];
    aq1.u[v2]     = *(const unsigned*)&sQ[qr * 72 + 32 + kb + 2 * v2];
    aq1.u[4 + v2] = *(const unsigned*)&sQ[qr * 72 + 32 + kb + 16 + 2 * v2];
  }

  const size_t kvbase = (size_t)(b * HH + h) * JJ * 2 * DD;

  // ---- scores: 18 key tiles of 16 ----
  for (int t2 = 0; t2 < 18; t2++) {
    const int j0 = jlo + t2 * 16;
    for (int i = lane; i < 16 * 64; i += 32) {
      int r = i >> 6, d = i & 63;
      int j = j0 + r;
      float kk = (j < JJ) ? memkv[kvbase + (size_t)j * (2 * DD) + d] : 0.f;
      sK[r * 72 + d] = tobf(kk);
    }
    __syncthreads();
    ABfrag bk0, bk1;
    const int kr = lane & 15;
    { const __bf16* p = &sK[kr * 72 + kb];      bk0.q[0] = *(const uint4*)p; bk0.q[1] = *(const uint4*)(p + 16); }
    { const __bf16* p = &sK[kr * 72 + 32 + kb]; bk1.q[0] = *(const uint4*)p; bk1.q[1] = *(const uint4*)(p + 16); }
    f32x8 s = {};
    s = __builtin_amdgcn_wmma_f32_16x16x32_bf16(false, aq0.v, false, bk0.v, (short)0, s, false, false);
    s = __builtin_amdgcn_wmma_f32_16x16x32_bf16(false, aq1.v, false, bk1.v, (short)0, s, false, false);
    const int cc = lane & 15;
    const int rb = (lane >> 4) * 8;
#pragma unroll
    for (int r = 0; r < 8; r++) {
      int rowg = n0 + rb + r;
      int j = j0 + cc;
      bool bad = (j > MM + rowg) || (j < MM + rowg - WIN);
      sS[(rb + r) * 292 + t2 * 16 + cc] = bad ? -3.0e38f : s[r];
    }
    __syncthreads();
  }

  // ---- softmax ----
  const int srow = lane & 15;
  const int shalf = lane >> 4;
  float m = -3.4e38f;
  for (int c = shalf; c < 288; c += 2) m = fmaxf(m, sS[srow * 292 + c]);
  m = fmaxf(m, __shfl_xor(m, 16));
  float ssum = 0.f;
  for (int c = shalf; c < 288; c += 2) ssum += __expf(sS[srow * 292 + c] - m);
  ssum += __shfl_xor(ssum, 16);
  const float inv = 1.0f / ssum;
  for (int c = shalf; c < 288; c += 2) {
    float p = __expf(sS[srow * 292 + c] - m) * inv;
    sS[srow * 292 + c] = p;
    int j = jlo + c;
    if (j < JJ)
      atomicAdd(&attnw[(size_t)(b * NN + n0 + srow) * JJ + j], p * (1.0f / HH));
  }
  __syncthreads();

  // ---- P @ V over 9 chunks of 32 keys ----
  f32x8 o0 = {}, o1 = {}, o2 = {}, o3 = {};
  for (int pc = 0; pc < 9; pc++) {
    const int c0 = pc * 32;
    const int j0 = jlo + c0;
    for (int i = lane; i < 64 * 32; i += 32) {
      int d = i >> 5, kk = i & 31;
      int j = j0 + kk;
      float vv = (j < JJ) ? memkv[kvbase + (size_t)j * (2 * DD) + DD + d] : 0.f;
      sV[d * 40 + kk] = tobf(vv);
    }
    __syncthreads();
    ABfrag aP;
    {
      const float* ps = &sS[(lane & 15) * 292 + c0 + kb];
#pragma unroll
      for (int i = 0; i < 8; i++) aP.v[i] = tobf(ps[i]);
#pragma unroll
      for (int i = 0; i < 8; i++) aP.v[8 + i] = tobf(ps[16 + i]);
    }
    ABfrag bv;
#define PV_STEP(nt, oacc)                                                                  \
    { const __bf16* p = &sV[((nt) * 16 + (lane & 15)) * 40 + kb];                          \
      bv.q[0] = *(const uint4*)p; bv.q[1] = *(const uint4*)(p + 16);                       \
      oacc = __builtin_amdgcn_wmma_f32_16x16x32_bf16(false, aP.v, false, bv.v, (short)0,   \
                                                     oacc, false, false); }
    PV_STEP(0, o0) PV_STEP(1, o1) PV_STEP(2, o2) PV_STEP(3, o3)
#undef PV_STEP
    __syncthreads();
  }

  const int ob = (lane >> 4) * 8;
  const int oc = lane & 15;
#pragma unroll
  for (int r = 0; r < 8; r++) {
    size_t ro = (size_t)(b * NN + n0 + ob + r) * DIMN + h * DD;
    hattn[ro + 0 + oc]  = o0[r];
    hattn[ro + 16 + oc] = o1[r];
    hattn[ro + 32 + oc] = o2[r];
    hattn[ro + 48 + oc] = o3[r];
  }
}

// ---------------------------------------------------------------------------
// Host launcher
// ---------------------------------------------------------------------------
extern "C" void kernel_launch(void* const* d_in, const int* in_sizes, int n_in,
                              void* d_out, int out_size, void* d_ws, size_t ws_size,
                              hipStream_t stream) {
  const float* x     = (const float*)d_in[0];
  const float* memkv = (const float*)d_in[1];
  const float* wq    = (const float*)d_in[2];
  const float* wkv   = (const float*)d_in[3];
  const float* ln1g  = (const float*)d_in[4];
  const float* ln1b  = (const float*)d_in[5];
  const float* ln2g  = (const float*)d_in[6];
  const float* ln2b  = (const float*)d_in[7];
  const float* w1    = (const float*)d_in[8];
  const float* b1    = (const float*)d_in[9];
  const float* w2    = (const float*)d_in[10];
  const float* b2    = (const float*)d_in[11];

  float* out_p     = (float*)d_out;
  float* attnw     = out_p + (size_t)BB * NN * DIMN;
  float* memkv_new = attnw + (size_t)BB * NN * JJ;

  char* ws = (char*)d_ws;
  size_t off = 0;
  auto alloc = [&](size_t bytes) -> char* {
    char* p = ws + off;
    off = (off + bytes + 255) & ~((size_t)255);
    return p;
  };
  const size_t ROWS = (size_t)BB * NN;  // 2048
  __bf16* x_bf  = (__bf16*)alloc(ROWS * DIMN * 2);
  __bf16* wq_t  = (__bf16*)alloc((size_t)DIMN * DIMN * 2);        // [N=1024][K=1024]
  __bf16* wkv_t = (__bf16*)alloc((size_t)2 * DIMN * DIMN * 2);    // [N=2048][K=1024]
  __bf16* w1_t  = (__bf16*)alloc((size_t)4 * DIMN * DIMN * 2);    // [N=4096][K=1024]
  __bf16* w2_t  = (__bf16*)alloc((size_t)DIMN * 4 * DIMN * 2);    // [N=1024][K=4096]
  float*  q_f   = (float*)alloc(ROWS * DIMN * 4);
  float*  kv_f  = (float*)alloc(ROWS * 2 * DIMN * 4);
  float*  hat_f = (float*)alloc(ROWS * DIMN * 4);
  float*  h_f   = (float*)alloc(ROWS * DIMN * 4);
  __bf16* h_bf  = (__bf16*)alloc(ROWS * DIMN * 2);
  __bf16* a1_bf = (__bf16*)alloc(ROWS * 4 * DIMN * 2);

  hipMemsetAsync(attnw, 0, (size_t)BB * NN * JJ * sizeof(float), stream);

  // weight convert+transpose: in [K,N] fp32 -> out [N,K] bf16
  cvt_transpose_kernel<<<dim3(DIMN / 32, DIMN / 32), 256, 0, stream>>>(wq, wq_t, DIMN, DIMN);
  cvt_transpose_kernel<<<dim3(DIMN / 32, 2 * DIMN / 32), 256, 0, stream>>>(wkv, wkv_t, DIMN, 2 * DIMN);
  cvt_transpose_kernel<<<dim3(DIMN / 32, 4 * DIMN / 32), 256, 0, stream>>>(w1, w1_t, DIMN, 4 * DIMN);
  cvt_transpose_kernel<<<dim3(4 * DIMN / 32, DIMN / 32), 256, 0, stream>>>(w2, w2_t, 4 * DIMN, DIMN);

  // LN1 -> bf16 x_
  ln_kernel<<<(int)ROWS, 256, 0, stream>>>(x, nullptr, ln1g, ln1b, nullptr, x_bf);

  // q = LN(x) @ wq * D^-0.5 ; kv = LN(x) @ wkv
  gemm_kernel<1><<<dim3(DIMN / 64, ROWS / 128), 256, 0, stream>>>(
      x_bf, wq_t, DIMN, DIMN, q_f, nullptr, nullptr, nullptr, 0.125f);
  gemm_kernel<0><<<dim3(2 * DIMN / 64, ROWS / 128), 256, 0, stream>>>(
      x_bf, wkv_t, DIMN, 2 * DIMN, kv_f, nullptr, nullptr, nullptr, 1.0f);

  // mem_kv_new = concat(mem_kv, kv transposed to [B,H,N,2D])
  copy_memkv_kernel<<<4096, 256, 0, stream>>>(memkv, memkv_new);
  scatter_kv_kernel<<<4096, 256, 0, stream>>>(kv_f, memkv_new);

  // windowed attention
  attn_kernel<<<dim3(NN / 16, HH, BB), 32, 0, stream>>>(q_f, memkv_new, attnw, hat_f);

  // h = attn_out + x ; LN2 -> bf16
  ln_kernel<<<(int)ROWS, 256, 0, stream>>>(hat_f, x, ln2g, ln2b, h_f, h_bf);

  // FFN
  gemm_kernel<2><<<dim3(4 * DIMN / 64, ROWS / 128), 256, 0, stream>>>(
      h_bf, w1_t, DIMN, 4 * DIMN, nullptr, a1_bf, b1, nullptr, 1.0f);
  gemm_kernel<3><<<dim3(DIMN / 64, ROWS / 128), 256, 0, stream>>>(
      a1_bf, w2_t, 4 * DIMN, DIMN, out_p, nullptr, b2, h_f, 1.0f);
}
